// AttnBlock3d_79139067396689
// MI455X (gfx1250) — matrix operand order repaired and checked by reading, not compiled, not run on hardware
//
#include <hip/hip_runtime.h>
#include <hip/hip_bf16.h>
#include <math.h>
#include <stdint.h>

typedef __attribute__((ext_vector_type(16))) _Float16 v16h;
typedef __attribute__((ext_vector_type(8)))  _Float16 v8h;
typedef __attribute__((ext_vector_type(8)))  float    v8f;

static constexpr int Bc = 4;
static constexpr int Cc = 256;
static constexpr int Nn = 4096;          // H*W*D = 16*16*16
static constexpr float SCALE = 0.0625f;  // C^-0.5 = 1/16

// ---------------------------------------------------------------------------
// WMMA fragment loaders (layouts per CDNA5 ISA 7.12.2, wave32)
// A (16x32 f16): lane L -> row m = L%16; v16h elems 0..7  = K[hb*8 .. hb*8+7],
//                elems 8..15 = K[16+hb*8 .. 16+hb*8+7]   (hb = L>>4)
// B (32x16 f16): lane L -> col n = L%16; same K order as A (elems 0..7 = K[hb*8..], 8..15 = K[16+hb*8..])
//                (source stored as B^T row-major: [n][K], contiguous)
// ---------------------------------------------------------------------------
__device__ inline v16h load_a_frag(const _Float16* row, int kk, int hb) {
    union { v16h v; v8h h[2]; } u;
    const _Float16* p = row + kk + hb * 8;
    u.h[0] = *(const v8h*)(p);
    u.h[1] = *(const v8h*)(p + 16);
    return u.v;
}
__device__ inline v16h load_b_frag(const _Float16* btrow, int kk, int hb) {
    union { v16h v; v8h h[2]; } u;
    const _Float16* p = btrow + kk + hb * 8;
    u.h[0] = *(const v8h*)(p);
    u.h[1] = *(const v8h*)(p + 16);
    return u.v;
}
__device__ inline v8f wmma_f16(v16h a, v16h b, v8f c) {
    return __builtin_amdgcn_wmma_f32_16x16x32_f16(false, a, false, b, (short)0, c,
                                                  false, false);
}

// CDNA5 async global->LDS copy (16 B per lane), tracked with ASYNCcnt.
// vdst = LDS byte address, vaddr = 64-bit global address (saddr = off).
__device__ inline void async_load_b128(unsigned lds_byte_addr, const void* gsrc) {
    asm volatile("global_load_async_to_lds_b128 %0, %1, off"
                 :: "v"(lds_byte_addr), "v"(gsrc)
                 : "memory");
}
// async loads complete in order: waiting <=8 guarantees the previous tile
// (8 loads/thread) has fully landed while the next tile's 8 stay in flight.
__device__ inline void wait_async8() {
    asm volatile("s_wait_asynccnt 0x8" ::: "memory");
}
__device__ inline void wait_async0() {
    asm volatile("s_wait_asynccnt 0x0" ::: "memory");
}
__device__ inline unsigned lds_addr_of(const void* p) {
    // flat LDS address: low 32 bits are the wave-relative LDS byte address
    return (unsigned)(uintptr_t)p;
}

// ---------------------------------------------------------------------------
// 1) GroupNorm(1 group) stats per batch: mean, rstd
// ---------------------------------------------------------------------------
__global__ __launch_bounds__(256) void stats_kernel(const float* __restrict__ x,
                                                    float* __restrict__ stats) {
    const int b = blockIdx.x;
    const size_t CN = (size_t)Cc * Nn;
    const float* xb = x + (size_t)b * CN;
    float s = 0.f, s2 = 0.f;
    for (size_t i = threadIdx.x; i < CN; i += 256) {
        float v = xb[i];
        s += v; s2 += v * v;
    }
    __shared__ float sh[512];
    sh[threadIdx.x] = s; sh[256 + threadIdx.x] = s2;
    __syncthreads();
    for (int o = 128; o > 0; o >>= 1) {
        if ((int)threadIdx.x < o) {
            sh[threadIdx.x]       += sh[threadIdx.x + o];
            sh[256 + threadIdx.x] += sh[256 + threadIdx.x + o];
        }
        __syncthreads();
    }
    if (threadIdx.x == 0) {
        float mean = sh[0] / (float)CN;
        float var  = sh[256] / (float)CN - mean * mean;
        stats[b * 2 + 0] = mean;
        stats[b * 2 + 1] = rsqrtf(var + 1e-6f);
    }
}

// ---------------------------------------------------------------------------
// 2) Normalize + affine + cast f16, transpose [B][C][N] -> [B][N][C]
// ---------------------------------------------------------------------------
__global__ __launch_bounds__(256) void norm_cast_kernel(
    const float* __restrict__ x, const float* __restrict__ stats,
    const float* __restrict__ gamma, const float* __restrict__ beta,
    _Float16* __restrict__ xn) {
    size_t idx = (size_t)blockIdx.x * 256 + threadIdx.x;  // b*N*C + n*C + c
    int c = (int)(idx & (Cc - 1));
    size_t t = idx >> 8;
    int n = (int)(t & (Nn - 1));
    int b = (int)(t >> 12);
    float mean = stats[b * 2 + 0], rstd = stats[b * 2 + 1];
    float v = (x[((size_t)b * Cc + c) * Nn + n] - mean) * rstd * gamma[c] + beta[c];
    xn[idx] = (_Float16)v;
}

__global__ __launch_bounds__(256) void cast_kernel(const float* __restrict__ s,
                                                   _Float16* __restrict__ d, int n) {
    int i = blockIdx.x * 256 + threadIdx.x;
    if (i < n) d[i] = (_Float16)s[i];
}

// ---------------------------------------------------------------------------
// 3) Generic WMMA GEMM with 2x2 register blocking:
//    Out = A(MxK) * Bt(NxK)^T, K = 256 (8 wmma k-steps), f16 in, f16 out.
//    Each wave computes a 32x32 output block (4 tiles) -> 2 b128 loads / wmma.
//    bias_mode: 0 -> bias[col], 1 -> bias[row], 2 -> none
// ---------------------------------------------------------------------------
__global__ __launch_bounds__(256) void gemm_nt_kernel(
    const _Float16* __restrict__ A, size_t aStride, int lda,
    const _Float16* __restrict__ Bt, size_t bStride, int ldb,
    _Float16* __restrict__ Co, size_t cStride, int ldc,
    int tilesN2, const float* __restrict__ bias, int bias_mode) {
    const int wave = threadIdx.x >> 5;
    const int lane = threadIdx.x & 31;
    const int ln = lane & 15, hb = lane >> 4;
    const int tile = blockIdx.x * 8 + wave;
    const int tm = tile / tilesN2, tn = tile % tilesN2;  // 32x32 blocks
    const int b = blockIdx.y;

    const _Float16* a0 = A + (size_t)b * aStride + (size_t)(tm * 32 + ln) * lda;
    const _Float16* a1 = a0 + (size_t)16 * lda;
    const _Float16* b0 = Bt + (size_t)b * bStride + (size_t)(tn * 32 + ln) * ldb;
    const _Float16* b1 = b0 + (size_t)16 * ldb;

    v8f acc00 = {}, acc01 = {}, acc10 = {}, acc11 = {};
#pragma unroll
    for (int s = 0; s < 8; ++s) {
        v16h fa0 = load_a_frag(a0, s * 32, hb);
        v16h fa1 = load_a_frag(a1, s * 32, hb);
        v16h fb0 = load_b_frag(b0, s * 32, hb);
        v16h fb1 = load_b_frag(b1, s * 32, hb);
        acc00 = wmma_f16(fa0, fb0, acc00);
        acc01 = wmma_f16(fa0, fb1, acc01);
        acc10 = wmma_f16(fa1, fb0, acc10);
        acc11 = wmma_f16(fa1, fb1, acc11);
    }
    _Float16* crow = Co + (size_t)b * cStride;
#pragma unroll
    for (int im = 0; im < 2; ++im) {
        v8f* accs = im ? &acc10 : &acc00;
#pragma unroll
        for (int in = 0; in < 2; ++in) {
            v8f acc = accs[in];
#pragma unroll
            for (int j = 0; j < 8; ++j) {
                int row = tm * 32 + im * 16 + j + 8 * hb;
                int col = tn * 32 + in * 16 + ln;
                float v = acc[j];
                if (bias_mode == 0) v += bias[col];
                else if (bias_mode == 1) v += bias[row];
                crow[(size_t)row * ldc + col] = (_Float16)v;
            }
        }
    }
}

// ---------------------------------------------------------------------------
// 4) Flash attention, double-buffered async pipeline.
//    block = 256 thr = 8 waves = 4 q-tiles(16 rows) x 2 column halves(128 ch).
//    K tile (32x256) and V tile (256x32, channel-major) in dynamic LDS,
//    2 buffers each; tile i+1 streams via ASYNCcnt while tile i computes.
//    Online softmax in f32; P re-shaped to A-fragment via per-wave LDS bounce.
// ---------------------------------------------------------------------------
static constexpr int LDK = 272;  // halfs, keeps 32B alignment, offsets banks
static constexpr int LDV = 48;
static constexpr int LDP = 40;
static constexpr int KS_HALFS = 32 * LDK;           // 8704
static constexpr int VS_HALFS = Cc * LDV;           // 12288
static constexpr int PS_HALFS = 8 * 16 * LDP;       // 5120
static constexpr unsigned FLASH_LDS_BYTES =
    2 * (2 * KS_HALFS + 2 * VS_HALFS) + 2 * PS_HALFS;  // 94208 B

__global__ __launch_bounds__(256, 1) void flash_kernel(
    const _Float16* __restrict__ Qp, const _Float16* __restrict__ Kp,
    const _Float16* __restrict__ Vc, _Float16* __restrict__ Hp) {
    extern __shared__ __attribute__((aligned(64))) _Float16 smem[];
    // NOTE: keep all LDS-derived pointers as scalars (never in initialized
    // arrays) -- constant-folded addrspacecast array initializers break lld.
    _Float16* Ks0 = smem;
    _Float16* Ks1 = smem + KS_HALFS;
    _Float16* Vs0 = smem + 2 * KS_HALFS;
    _Float16* Vs1 = smem + 2 * KS_HALFS + VS_HALFS;
    _Float16* Ps  = smem + 2 * KS_HALFS + 2 * VS_HALFS;

    const int tid = threadIdx.x;
    const int wave = tid >> 5, lane = tid & 31;
    const int ln = lane & 15, hb = lane >> 4;
    const int wm = wave >> 1;      // q tile within block
    const int wn = wave & 1;       // which 128-channel half of O
    const int b = blockIdx.y;
    const int q0 = blockIdx.x * 64 + wm * 16;

    const _Float16* Qb = Qp + (size_t)b * Nn * Cc;
    const _Float16* Kb = Kp + (size_t)b * Nn * Cc;
    const _Float16* Vb = Vc + (size_t)b * Cc * Nn;

    // per-thread staging addresses
    const int kr = tid >> 3, kcb = (tid & 7) * 32;          // K: row, col-base
    const _Float16* ksrc_row = Kb + (size_t)kr * Cc + kcb;  // + kt*Cc later
    const _Float16* vsrc_row = Vb + (size_t)tid * Nn;       // + kt later
    const unsigned kdst0 = lds_addr_of(Ks0 + kr * LDK + kcb);
    const unsigned kdst1 = lds_addr_of(Ks1 + kr * LDK + kcb);
    const unsigned vdst0 = lds_addr_of(Vs0 + tid * LDV);
    const unsigned vdst1 = lds_addr_of(Vs1 + tid * LDV);

    auto issue_tile = [&](int kt, int buf) {
        const _Float16* ks = ksrc_row + (size_t)kt * Cc;
        const _Float16* vs = vsrc_row + kt;
        const unsigned kd = buf ? kdst1 : kdst0;
        const unsigned vd = buf ? vdst1 : vdst0;
#pragma unroll
        for (int j = 0; j < 4; ++j) async_load_b128(kd + j * 16, ks + j * 8);
#pragma unroll
        for (int j = 0; j < 4; ++j) async_load_b128(vd + j * 16, vs + j * 8);
    };

    // Q fragments live in registers for the whole pass (reused 128x)
    v16h qa[8];
    {
        const _Float16* qrow = Qb + (size_t)(q0 + ln) * Cc;
#pragma unroll
        for (int s = 0; s < 8; ++s) qa[s] = load_a_frag(qrow, s * 32, hb);
    }

    v8f o[8];
#pragma unroll
    for (int t = 0; t < 8; ++t)
#pragma unroll
        for (int j = 0; j < 8; ++j) o[t][j] = 0.f;
    float mrow[8], lrow[8];
#pragma unroll
    for (int j = 0; j < 8; ++j) { mrow[j] = -1e30f; lrow[j] = 0.f; }

    _Float16* Pw = Ps + wave * 16 * LDP;

    issue_tile(0, 0);  // preload first tile

    const int nIter = Nn / 32;
    for (int i = 0; i < nIter; ++i) {
        const int cur = i & 1;
        const int ktn = (i + 1) * 32;
        if (ktn < Nn) {
            // stream next tile into the other buffer while we compute this one
            issue_tile(ktn, cur ^ 1);
            if (ktn + 32 < Nn) {  // nudge tile i+2 toward L2
                __builtin_prefetch(ksrc_row + (size_t)(ktn + 32) * Cc, 0, 3);
                __builtin_prefetch(vsrc_row + ktn + 32, 0, 3);
            }
            wait_async8();   // tile i (first 8 loads) has landed
        } else {
            wait_async0();
        }
        __syncthreads();     // all waves' slices of tile i are in LDS

        const _Float16* Ks = cur ? Ks1 : Ks0;
        const _Float16* Vs = cur ? Vs1 : Vs0;

        // S = Q K^T for this 32-key tile: two 16x16 f32 fragments
        v8f s0, s1;
#pragma unroll
        for (int j = 0; j < 8; ++j) { s0[j] = 0.f; s1[j] = 0.f; }
        const _Float16* k0row = Ks + ln * LDK;
        const _Float16* k1row = Ks + (16 + ln) * LDK;
#pragma unroll
        for (int s = 0; s < 8; ++s) {
            v16h fb0 = load_b_frag(k0row, s * 32, hb);
            v16h fb1 = load_b_frag(k1row, s * 32, hb);
            s0 = wmma_f16(qa[s], fb0, s0);
            s1 = wmma_f16(qa[s], fb1, s1);
        }

        // online softmax; row r = j + 8*hb lives in one 16-lane group at vgpr j
        float al[8];
#pragma unroll
        for (int j = 0; j < 8; ++j) {
            float v0 = s0[j] * SCALE, v1 = s1[j] * SCALE;
            float rm = fmaxf(v0, v1);
            rm = fmaxf(rm, __shfl_xor(rm, 1, 16));
            rm = fmaxf(rm, __shfl_xor(rm, 2, 16));
            rm = fmaxf(rm, __shfl_xor(rm, 4, 16));
            rm = fmaxf(rm, __shfl_xor(rm, 8, 16));
            float mn = fmaxf(mrow[j], rm);
            float a = __expf(mrow[j] - mn);
            float p0 = __expf(v0 - mn);
            float p1 = __expf(v1 - mn);
            float rs = p0 + p1;
            rs += __shfl_xor(rs, 1, 16);
            rs += __shfl_xor(rs, 2, 16);
            rs += __shfl_xor(rs, 4, 16);
            rs += __shfl_xor(rs, 8, 16);
            lrow[j] = lrow[j] * a + rs;
            mrow[j] = mn;
            al[j] = a;
            int row = j + 8 * hb;
            Pw[row * LDP + ln]      = (_Float16)p0;
            Pw[row * LDP + 16 + ln] = (_Float16)p1;
        }
        // rescale O accumulators
#pragma unroll
        for (int t = 0; t < 8; ++t)
#pragma unroll
            for (int j = 0; j < 8; ++j) o[t][j] *= al[j];

        // P back as A fragment (same-wave LDS RAW; compiler inserts ds waits)
        v16h pa = load_a_frag(Pw + (size_t)ln * LDP, 0, hb);

        // O += P * V  (this wave covers channels wn*128 .. wn*128+127)
#pragma unroll
        for (int t = 0; t < 8; ++t) {
            int ct = wn * 8 + t;
            v16h bv = load_b_frag(Vs + (ct * 16 + ln) * LDV, 0, hb);
            o[t] = wmma_f16(pa, bv, o[t]);
        }
        __syncthreads();  // done reading buf[cur]; safe to refill at i+2
    }

    // epilogue: O /= l, store pos-major f16
    _Float16* Hb = Hp + (size_t)b * Nn * Cc;
#pragma unroll
    for (int j = 0; j < 8; ++j) {
        float inv = 1.0f / lrow[j];
        int row = q0 + j + 8 * hb;
#pragma unroll
        for (int t = 0; t < 8; ++t) {
            int col = (wn * 8 + t) * 16 + ln;
            Hb[(size_t)row * Cc + col] = (_Float16)(o[t][j] * inv);
        }
    }
}

// ---------------------------------------------------------------------------
// 5) Projection + bias + residual (2x2 blocked): out[b][c][n] = x + proj_w @ h
// ---------------------------------------------------------------------------
__global__ __launch_bounds__(256) void proj_kernel(
    const _Float16* __restrict__ Hp, const _Float16* __restrict__ Wp,
    const float* __restrict__ x, const float* __restrict__ pb,
    float* __restrict__ out) {
    const int wave = threadIdx.x >> 5, lane = threadIdx.x & 31;
    const int ln = lane & 15, hb = lane >> 4;
    const int tile = blockIdx.x * 8 + wave;
    const int tm = tile >> 3, tn = tile & 7;  // tm: 128 pos blocks, tn: 8 ch blocks
    const int b = blockIdx.y;

    const _Float16* a0 = Hp + (size_t)b * Nn * Cc + (size_t)(tm * 32 + ln) * Cc;
    const _Float16* a1 = a0 + (size_t)16 * Cc;
    const _Float16* b0 = Wp + (size_t)(tn * 32 + ln) * Cc;
    const _Float16* b1 = b0 + (size_t)16 * Cc;

    v8f acc00 = {}, acc01 = {}, acc10 = {}, acc11 = {};
#pragma unroll
    for (int s = 0; s < 8; ++s) {
        v16h fa0 = load_a_frag(a0, s * 32, hb);
        v16h fa1 = load_a_frag(a1, s * 32, hb);
        v16h fb0 = load_b_frag(b0, s * 32, hb);
        v16h fb1 = load_b_frag(b1, s * 32, hb);
        acc00 = wmma_f16(fa0, fb0, acc00);
        acc01 = wmma_f16(fa0, fb1, acc01);
        acc10 = wmma_f16(fa1, fb0, acc10);
        acc11 = wmma_f16(fa1, fb1, acc11);
    }
#pragma unroll
    for (int im = 0; im < 2; ++im) {
        v8f* accs = im ? &acc10 : &acc00;
#pragma unroll
        for (int in = 0; in < 2; ++in) {
            v8f acc = accs[in];
#pragma unroll
            for (int j = 0; j < 8; ++j) {
                int n = tm * 32 + im * 16 + j + 8 * hb;  // position
                int c = tn * 32 + in * 16 + ln;          // channel
                size_t idx = ((size_t)b * Cc + c) * Nn + n;
                out[idx] = x[idx] + pb[c] + acc[j];
            }
        }
    }
}

// ---------------------------------------------------------------------------
extern "C" void kernel_launch(void* const* d_in, const int* in_sizes, int n_in,
                              void* d_out, int out_size, void* d_ws, size_t ws_size,
                              hipStream_t stream) {
    (void)in_sizes; (void)n_in; (void)out_size; (void)ws_size;
    const float* x      = (const float*)d_in[0];
    const float* gamma  = (const float*)d_in[1];
    const float* beta   = (const float*)d_in[2];
    const float* qkv_w  = (const float*)d_in[3];
    const float* qkv_b  = (const float*)d_in[4];
    const float* proj_w = (const float*)d_in[5];
    const float* proj_b = (const float*)d_in[6];
    float* out = (float*)d_out;

    const size_t BNC = (size_t)Bc * Nn * Cc;  // per-tensor f16 element count
    char* w = (char*)d_ws;
    float*     stats = (float*)w;                         // 256 B slot
    _Float16*  Xn    = (_Float16*)(w + 256);
    _Float16*  Qp    = Xn + BNC;
    _Float16*  Kp    = Qp + BNC;
    _Float16*  Vc    = Kp + BNC;
    _Float16*  Hp    = Vc + BNC;
    _Float16*  Wqkv  = Hp + BNC;                          // 768x256
    _Float16*  Wproj = Wqkv + (size_t)768 * 256;          // 256x256

    stats_kernel<<<Bc, 256, 0, stream>>>(x, stats);
    norm_cast_kernel<<<(int)(BNC / 256), 256, 0, stream>>>(x, stats, gamma, beta, Xn);
    cast_kernel<<<(768 * 256) / 256, 256, 0, stream>>>(qkv_w, Wqkv, 768 * 256);
    cast_kernel<<<(256 * 256) / 256, 256, 0, stream>>>(proj_w, Wproj, 256 * 256);

    const size_t NC = (size_t)Nn * Cc;
    // Q = Xn * Wq^T        [N x 256] pos-major; 1024 32x32 blocks/batch
    gemm_nt_kernel<<<dim3(128, Bc), 256, 0, stream>>>(
        Xn, NC, Cc, Wqkv, 0, Cc, Qp, NC, Cc, 8, qkv_b, 0);
    // K = Xn * Wk^T        [N x 256] pos-major
    gemm_nt_kernel<<<dim3(128, Bc), 256, 0, stream>>>(
        Xn, NC, Cc, Wqkv + (size_t)256 * 256, 0, Cc, Kp, NC, Cc, 8, qkv_b + 256, 0);
    // V = Wv * Xn^T        [256 x N] channel-major
    gemm_nt_kernel<<<dim3(128, Bc), 256, 0, stream>>>(
        Wqkv + (size_t)512 * 256, 0, Cc, Xn, NC, Cc, Vc, NC, Nn, 128, qkv_b + 512, 1);

    flash_kernel<<<dim3(Nn / 64, Bc), 256, FLASH_LDS_BYTES, stream>>>(Qp, Kp, Vc, Hp);

    proj_kernel<<<dim3(128, Bc), 256, 0, stream>>>(Hp, Wproj, x, proj_b, out);
}
